// GRU_model_63488206569534
// MI455X (gfx1250) — compile-verified, matrix operand order checked by reading, not run on hardware
//
#include <hip/hip_runtime.h>
#include <hip/hip_bf16.h>

// ---------------- problem constants ----------------
#define BB   32
#define TT   1024
#define II   256
#define HH   512
#define G3   1536            // 3*H
#define NWG  8               // workgroups in recurrence
#define COLS 64              // hidden columns per workgroup (8*64 = 512)
#define TCHUNK 128           // timesteps per x_proj chunk

// ---------------- workspace layout (bytes) ----------------
#define WS_XPROJ   0u                      // [TCHUNK][B][3H] f32   = 25165824
#define WS_WHH     25165824u               // [1536][512] f16       = 1572864
#define WS_WIH     26738688u               // [1536][256] f16       = 786432
#define WS_X16     27525120u               // [32][1024][256] f16   = 16777216
#define WS_H0F     44302336u               // [32][512] f32         = 65536
#define WS_H1F     44367872u               // [32][512] f32         = 65536
#define WS_H0H     44433408u               // [32][512] f16         = 32768
#define WS_H1H     44466176u               // [32][512] f16         = 32768
#define WS_SYNC    44498944u               // counter

typedef __attribute__((ext_vector_type(16))) _Float16 v16h;
typedef __attribute__((ext_vector_type(8)))  _Float16 v8h;
typedef __attribute__((ext_vector_type(8)))  float    v8f;

// ---- WMMA wrapper (v_wmma_f32_16x16x32_f16) ----
__device__ __forceinline__ v8f wmma16x16x32(v16h a, v16h b, v8f c) {
  return __builtin_amdgcn_wmma_f32_16x16x32_f16(
      /*neg_a=*/false, a, /*neg_b=*/false, b,
      /*c_mod=*/(short)0, c, /*reuse_a=*/false, /*reuse_b=*/false);
}

// 16x32 f16 fragment per ISA 16-bit layout, from an f16 row (LDS or global):
// lane<16 holds K={kb..kb+7, kb+16..kb+23}; lane>=16 holds K={+8..+15, +24..+31}.
__device__ __forceinline__ v16h frag_h16(const _Float16* p, int half) {
  v8h c0 = *(const v8h*)(p + half * 8);
  v8h c1 = *(const v8h*)(p + 16 + half * 8);
  v16h f;
#pragma unroll
  for (int i = 0; i < 8; ++i) { f[i] = c0[i]; f[8 + i] = c1[i]; }
  return f;
}

__device__ __forceinline__ float sigmoidf_(float x) {
  return 1.0f / (1.0f + __expf(-x));
}

// ---------------- kernel 0: one-time f16 packing + state init ---------------
__global__ void gru_init_pack(const float* __restrict__ w_hh,
                              const float* __restrict__ w_ih,
                              const float* __restrict__ x,
                              _Float16* __restrict__ whh16,
                              _Float16* __restrict__ wih16,
                              _Float16* __restrict__ x16,
                              float* __restrict__ h0f, float* __restrict__ h1f,
                              _Float16* __restrict__ h0h, _Float16* __restrict__ h1h,
                              unsigned int* __restrict__ sync) {
  const size_t i = (size_t)blockIdx.x * blockDim.x + threadIdx.x;
  const size_t stride = (size_t)gridDim.x * blockDim.x;
  for (size_t k = i; k < (size_t)BB * TT * II; k += stride) x16[k] = (_Float16)x[k];
  for (size_t k = i; k < (size_t)G3 * HH; k += stride) whh16[k] = (_Float16)w_hh[k];
  for (size_t k = i; k < (size_t)G3 * II; k += stride) wih16[k] = (_Float16)w_ih[k];
  for (size_t k = i; k < (size_t)BB * HH; k += stride) {
    h0f[k] = 0.0f; h1f[k] = 0.0f;
    h0h[k] = (_Float16)0.0f; h1h[k] = (_Float16)0.0f;
  }
  if (i == 0) *sync = 0u;
}

// ---------------- kernel 1: x_proj chunk GEMM (all-f16 operands) ------------
// xp[tl][b][g] = sum_i x[b, t_base+tl, i] * w_ih[g, i] + b_ih[g]
// M = TCHUNK*B = 4096 (row = tl*32 + b), N = 1536, K = 256. Wave = 16x64 strip.
__global__ void gru_xproj_gemm(const _Float16* __restrict__ x16,
                               const _Float16* __restrict__ wih16,
                               const float* __restrict__ b_ih,
                               float* __restrict__ xp,
                               int t_base) {
  const int tid  = threadIdx.x;
  const int ln   = tid & 31;
  const int li   = ln & 15;
  const int half = ln >> 4;
  const int gw   = blockIdx.x * (blockDim.x >> 5) + (tid >> 5);
  const int mtile = gw / 24;          // 0..255
  const int strip = gw % 24;          // 0..23 -> 64 N columns
  const int m0 = mtile * 16;
  const int j0 = strip * 64;

  const int row = m0 + li;            // A row for this lane
  const int tl  = row >> 5;
  const int b   = row & 31;
  const _Float16* arow = x16 + ((size_t)b * TT + (t_base + tl)) * II;

  v8f acc0 = {}, acc1 = {}, acc2 = {}, acc3 = {};
#pragma unroll
  for (int kb = 0; kb < II; kb += 32) {
    v16h a = frag_h16(arow + kb, half);
    acc0 = wmma16x16x32(a, frag_h16(wih16 + (size_t)(j0 +  0 + li) * II + kb, half), acc0);
    acc1 = wmma16x16x32(a, frag_h16(wih16 + (size_t)(j0 + 16 + li) * II + kb, half), acc1);
    acc2 = wmma16x16x32(a, frag_h16(wih16 + (size_t)(j0 + 32 + li) * II + kb, half), acc2);
    acc3 = wmma16x16x32(a, frag_h16(wih16 + (size_t)(j0 + 48 + li) * II + kb, half), acc3);
  }
  v8f acc[4] = {acc0, acc1, acc2, acc3};
#pragma unroll
  for (int sub = 0; sub < 4; ++sub) {
    const int g = j0 + sub * 16 + li;
    const float bias = b_ih[g];
#pragma unroll
    for (int p = 0; p < 8; ++p) {
      const int r = m0 + half * 8 + p;         // = tl*32 + b ordering
      xp[(size_t)r * G3 + g] = acc[sub][p] + bias;
    }
  }
}

// ---------------- kernel 2: persistent GRU recurrence chunk -----------------
// NWG workgroups x 256 threads (8 waves). WG owns hidden columns
// [wg*64, wg*64+64) for all 3 gates; weights LDS-resident (192 KB, staged via
// global_load_async_to_lds_b128). h state: f32 (elementwise carry) + f16
// mirror (WMMA A operand), both ping-ponged; one grid sync per step.
__global__ void gru_recurrence_chunk(const float* __restrict__ xp,     // [TCHUNK][32][1536]
                                     const _Float16* __restrict__ whh, // [1536][512]
                                     const float* __restrict__ b_hh,
                                     float* __restrict__ h0f, float* __restrict__ h1f,
                                     _Float16* __restrict__ h0h, _Float16* __restrict__ h1h,
                                     unsigned int* __restrict__ sync,
                                     int round_base, int steps) {
  extern __shared__ _Float16 lw[];     // [3][COLS][512] f16 = 192 KB

  const int base = blockIdx.x * COLS;  // global column base of this WG
  // Stage this WG's weight slice into LDS with CDNA5 async copies (ASYNCcnt).
  for (int e = threadIdx.x; e < 3 * COLS * (HH / 8); e += blockDim.x) {
    const int k8 = e & 63;             // 64 x 16B chunks per 512-half row
    const int rj = e >> 6;             // 0..191 = g*COLS + j
    const int g  = rj / COLS;
    const int j  = rj % COLS;
    const _Float16* src = whh + (size_t)(g * HH + base + j) * HH + k8 * 8;
    const _Float16* dst = lw + (size_t)rj * HH + k8 * 8;
    unsigned lds_off = (unsigned)(unsigned long long)dst;
    unsigned long long gaddr = (unsigned long long)src;
    asm volatile("global_load_async_to_lds_b128 %0, %1, off"
                 :: "v"(lds_off), "v"(gaddr) : "memory");
  }
  asm volatile("s_wait_asynccnt 0" ::: "memory");
  __syncthreads();

  const int tid  = threadIdx.x;
  const int ln   = tid & 31;
  const int li   = ln & 15;
  const int half = ln >> 4;
  const int wv   = tid >> 5;           // 0..7
  const int m0   = (wv & 1) * 16;      // batch-row tile
  const int nt   = wv >> 1;            // 0..3 column tile
  const int jcol = nt * 16 + li;       // local column (B fragment lane)
  const int jglob = base + jcol;

  const float bhr = b_hh[jglob];
  const float bhz = b_hh[HH + jglob];
  const float bhn = b_hh[2 * HH + jglob];
  const _Float16* brow_r = lw + (size_t)(0 * COLS + jcol) * HH;
  const _Float16* brow_z = lw + (size_t)(1 * COLS + jcol) * HH;
  const _Float16* brow_n = lw + (size_t)(2 * COLS + jcol) * HH;

  for (int s = 0; s < steps; ++s) {
    const float*    hinf  = (s & 1) ? h1f : h0f;
    float*          houtf = (s & 1) ? h0f : h1f;
    const _Float16* hinh  = (s & 1) ? h1h : h0h;
    _Float16*       houth = (s & 1) ? h0h : h1h;

    // prefetch next step's x_proj row (global_prefetch_b8)
    {
      const int sn = (s + 1 < steps) ? (s + 1) : s;
      __builtin_prefetch(xp + (size_t)sn * BB * G3 + (m0 + half * 8) * G3 + jglob, 0, 3);
    }

    v8f accr = {}, accz = {}, accn = {};
    const _Float16* arow = hinh + (size_t)(m0 + li) * HH;
#pragma unroll 8
    for (int kb = 0; kb < HH; kb += 32) {
      v16h a = frag_h16(arow + kb, half);
      accr = wmma16x16x32(a, frag_h16(brow_r + kb, half), accr);
      accz = wmma16x16x32(a, frag_h16(brow_z + kb, half), accz);
      accn = wmma16x16x32(a, frag_h16(brow_n + kb, half), accn);
    }

    // elementwise gate math in f32 (C/D layout: lane holds N=jglob, M=half*8+p)
    const float* xrow = xp + (size_t)s * BB * G3;
#pragma unroll
    for (int p = 0; p < 8; ++p) {
      const int m = m0 + half * 8 + p;
      const float* xr = xrow + (size_t)m * G3 + jglob;
      const float r = sigmoidf_(xr[0]      + accr[p] + bhr);
      const float z = sigmoidf_(xr[HH]     + accz[p] + bhz);
      const float n = tanhf(xr[2 * HH] + r * (accn[p] + bhn));
      const float hp = hinf[(size_t)m * HH + jglob];
      const float hnew = (1.0f - z) * n + z * hp;
      houtf[(size_t)m * HH + jglob] = hnew;
      houth[(size_t)m * HH + jglob] = (_Float16)hnew;   // f16 mirror for next WMMA
    }

    // ---- grid sync across the NWG persistent workgroups ----
    __threadfence();
    __syncthreads();
    if (tid == 0) {
      atomicAdd(sync, 1u);
      const unsigned target = (unsigned)(NWG * (round_base + s + 1));
      while (__hip_atomic_load(sync, __ATOMIC_ACQUIRE, __HIP_MEMORY_SCOPE_AGENT) < target) {
        __builtin_amdgcn_s_sleep(2);
      }
    }
    __syncthreads();
  }
}

// ---------------- kernel 3: final FC  out[b] = h_last[b,:] . fc_w + fc_b ----
__global__ void gru_fc(const float* __restrict__ h,
                       const float* __restrict__ fc_w,
                       const float* __restrict__ fc_b,
                       float* __restrict__ out) {
  const int tid = threadIdx.x;     // 1024 = 32 waves, one wave per batch row
  const int b = tid >> 5;
  const int k = tid & 31;
  float s = 0.0f;
#pragma unroll
  for (int j = k; j < HH; j += 32) s += h[(size_t)b * HH + j] * fc_w[j];
#pragma unroll
  for (int off = 16; off > 0; off >>= 1) s += __shfl_xor(s, off, 32);
  if (k == 0) out[b] = s + fc_b[0];
}

// ---------------- host launch ----------------
extern "C" void kernel_launch(void* const* d_in, const int* in_sizes, int n_in,
                              void* d_out, int out_size, void* d_ws, size_t ws_size,
                              hipStream_t stream) {
  const float* x    = (const float*)d_in[0];   // [32,1024,256]
  const float* w_ih = (const float*)d_in[1];   // [1536,256]
  const float* w_hh = (const float*)d_in[2];   // [1536,512]
  const float* b_ih = (const float*)d_in[3];   // [1536]
  const float* b_hh = (const float*)d_in[4];   // [1536]
  const float* fc_w = (const float*)d_in[5];   // [512]
  const float* fc_b = (const float*)d_in[6];   // [1]
  float* out = (float*)d_out;                  // [32]

  char* ws = (char*)d_ws;
  float*        xp    = (float*)(ws + WS_XPROJ);
  _Float16*     whh16 = (_Float16*)(ws + WS_WHH);
  _Float16*     wih16 = (_Float16*)(ws + WS_WIH);
  _Float16*     x16   = (_Float16*)(ws + WS_X16);
  float*        h0f   = (float*)(ws + WS_H0F);
  float*        h1f   = (float*)(ws + WS_H1F);
  _Float16*     h0h   = (_Float16*)(ws + WS_H0H);
  _Float16*     h1h   = (_Float16*)(ws + WS_H1H);
  unsigned int* sync  = (unsigned int*)(ws + WS_SYNC);

  // 192 KB dynamic LDS for the weight-resident recurrence (CDNA5: 320 KB/WGP)
  (void)hipFuncSetAttribute(reinterpret_cast<const void*>(gru_recurrence_chunk),
                            hipFuncAttributeMaxDynamicSharedMemorySize,
                            3 * COLS * HH * (int)sizeof(_Float16));

  gru_init_pack<<<2048, 256, 0, stream>>>(w_hh, w_ih, x, whh16, wih16, x16,
                                          h0f, h1f, h0h, h1h, sync);

  for (int c = 0; c < TT / TCHUNK; ++c) {
    // 4096x1536x256 GEMM chunk: 256 mtiles * 24 strips = 6144 waves / 8 per block
    gru_xproj_gemm<<<768, 256, 0, stream>>>(x16, wih16, b_ih, xp, c * TCHUNK);
    gru_recurrence_chunk<<<NWG, 256, 3 * COLS * HH * sizeof(_Float16), stream>>>(
        xp, whh16, b_hh, h0f, h1f, h0h, h1h, sync, c * TCHUNK, TCHUNK);
  }

  // after every 128-step chunk the live h lands back in buffer 0
  gru_fc<<<1, 1024, 0, stream>>>(h0f, fc_w, fc_b, out);
}
